// XFormerDecoderLayer_24962349924418
// MI455X (gfx1250) — matrix operand order repaired and checked by reading, not compiled, not run on hardware
//
#include <hip/hip_runtime.h>
#include <hip/hip_bf16.h>
#include <math.h>

typedef __attribute__((ext_vector_type(16))) __bf16        v16bf;
typedef __attribute__((ext_vector_type(8)))  float         v8f;
typedef __attribute__((ext_vector_type(16))) unsigned short v16us;
typedef __attribute__((ext_vector_type(8)))  unsigned short v8us;

#define DEVI static __device__ __forceinline__

constexpr int   Bc = 2, Tc = 2048, Sc = 512, Dc = 1024, Hc = 16, Fc = 4096;
constexpr int   TSc = Tc + Sc;               // 2560 keys (self + ctx)
constexpr float QSCALE = 0.125f * 1.44269504088896340736f; // 1/sqrt(64) * log2(e)

DEVI unsigned short f2bf(float f) {
  unsigned u = __float_as_uint(f);
  u += 0x7FFFu + ((u >> 16) & 1u);           // round-to-nearest-even
  return (unsigned short)(u >> 16);
}

DEVI v16bf usbf(v16us v) { return __builtin_bit_cast(v16bf, v); }

DEVI v16us combine8(v8us lo, v8us hi) {
  return __builtin_shufflevector(lo, hi, 0,1,2,3,4,5,6,7,8,9,10,11,12,13,14,15);
}

DEVI v8f vzero8() {
  v8f v;
  #pragma unroll
  for (int i = 0; i < 8; ++i) v[i] = 0.0f;
  return v;
}

DEVI v8f wmma_bf16(v16bf a, v16bf b, v8f c) {
  return __builtin_amdgcn_wmma_f32_16x16x32_bf16(false, a, false, b, (short)0, c, false, false);
}

// Per-wave LDS ordering: DS ops of one wave are in-order; the wait + memory
// clobber stops compiler reordering of the P store->load round-trip.
DEVI void lds_fence_wave() { asm volatile("s_wait_dscnt 0x0" ::: "memory"); }

// ---------------------------------------------------------------- converts
__global__ void k_f32_to_bf16(const float* __restrict__ in,
                              unsigned short* __restrict__ out, int n) {
  int i = blockIdx.x * blockDim.x + threadIdx.x;
  int stride = gridDim.x * blockDim.x;
  for (; i < n; i += stride) out[i] = f2bf(in[i]);
}

// Transpose + convert: in f32 [K][N]  ->  out bf16 [N][K]   (K,N multiples of 32)
__global__ __launch_bounds__(256)
void k_wt_bf16(const float* __restrict__ in, unsigned short* __restrict__ out,
               int K, int N) {
  __shared__ unsigned short t[32][33];
  const int tx = threadIdx.x & 31, ty = threadIdx.x >> 5;  // 32 x 8
  const int n0 = blockIdx.x * 32, k0 = blockIdx.y * 32;
  #pragma unroll
  for (int i = 0; i < 4; ++i)
    t[ty + 8 * i][tx] = f2bf(in[(size_t)(k0 + ty + 8 * i) * N + n0 + tx]);
  __syncthreads();
  #pragma unroll
  for (int i = 0; i < 4; ++i)
    out[(size_t)(n0 + ty + 8 * i) * K + k0 + tx] = t[tx][ty + 8 * i];
}

// ---------------------------------------------------------------- reductions
DEVI void blk_red2(float& s, float& s2, float* red) {
  #pragma unroll
  for (int o = 16; o; o >>= 1) { s += __shfl_xor(s, o, 32); s2 += __shfl_xor(s2, o, 32); }
  const int wid = threadIdx.x >> 5;
  __syncthreads();                            // protect red from previous use
  if ((threadIdx.x & 31) == 0) { red[wid] = s; red[8 + wid] = s2; }
  __syncthreads();
  float a = 0.f, b = 0.f;
  #pragma unroll
  for (int i = 0; i < 8; ++i) { a += red[i]; b += red[8 + i]; }
  s = a; s2 = b;
}

// LN(x; g, b) -> bf16, one block per row of 1024
__global__ __launch_bounds__(256)
void k_ln1(const float* __restrict__ x, const float* __restrict__ g,
           const float* __restrict__ be, unsigned short* __restrict__ out) {
  __shared__ float red[16];
  const int row = blockIdx.x;
  const float* xr = x + (size_t)row * Dc;
  const int base = threadIdx.x * 4;
  float4 xv = *(const float4*)(xr + base);
  float v[4] = {xv.x, xv.y, xv.z, xv.w};
  float s = v[0] + v[1] + v[2] + v[3];
  float s2 = v[0]*v[0] + v[1]*v[1] + v[2]*v[2] + v[3]*v[3];
  blk_red2(s, s2, red);
  const float mean = s * (1.0f / Dc);
  const float var  = s2 * (1.0f / Dc) - mean * mean;
  const float rstd = rsqrtf(var + 1e-5f);
  #pragma unroll
  for (int i = 0; i < 4; ++i) {
    const int j = base + i;
    out[(size_t)row * Dc + j] = f2bf((v[i] - mean) * rstd * g[j] + be[j]);
  }
}

// LN(LN(x; g2, be2); gf, bf) -> bf16
__global__ __launch_bounds__(256)
void k_ln2(const float* __restrict__ x, const float* __restrict__ g2,
           const float* __restrict__ be2, const float* __restrict__ gf,
           const float* __restrict__ bfv, unsigned short* __restrict__ out) {
  __shared__ float red[16];
  const int row = blockIdx.x;
  const float* xr = x + (size_t)row * Dc;
  const int base = threadIdx.x * 4;
  float4 xv = *(const float4*)(xr + base);
  float v[4] = {xv.x, xv.y, xv.z, xv.w};
  float s = v[0] + v[1] + v[2] + v[3];
  float s2 = v[0]*v[0] + v[1]*v[1] + v[2]*v[2] + v[3]*v[3];
  blk_red2(s, s2, red);
  float mean = s * (1.0f / Dc);
  float var  = s2 * (1.0f / Dc) - mean * mean;
  float rstd = rsqrtf(var + 1e-5f);
  float y[4];
  s = 0.f; s2 = 0.f;
  #pragma unroll
  for (int i = 0; i < 4; ++i) {
    const int j = base + i;
    y[i] = (v[i] - mean) * rstd * g2[j] + be2[j];
    s += y[i]; s2 += y[i] * y[i];
  }
  blk_red2(s, s2, red);
  mean = s * (1.0f / Dc);
  var  = s2 * (1.0f / Dc) - mean * mean;
  rstd = rsqrtf(var + 1e-5f);
  #pragma unroll
  for (int i = 0; i < 4; ++i) {
    const int j = base + i;
    out[(size_t)row * Dc + j] = f2bf((y[i] - mean) * rstd * gf[j] + bfv[j]);
  }
}

// ---------------------------------------------------------------- WMMA GEMM
// C[M,N] = A[M,K](bf16,row) @ WT[N,K](bf16,row)^T + bias, epilogue per EPI.
// EPI 0: QKV scatter (q scaled; v stored dh-major)   EPI 1: CKV scatter
// EPI 2: GELU -> bf16                                EPI 3: + residual -> f32
template <int EPI>
__global__ __launch_bounds__(256)
void k_gemm(const unsigned short* __restrict__ A, const unsigned short* __restrict__ WT,
            const float* __restrict__ bias, int M, int N, int K,
            unsigned short* __restrict__ ou0, unsigned short* __restrict__ ou1,
            unsigned short* __restrict__ ou2, const float* __restrict__ addsrc,
            float* __restrict__ of32) {
  __shared__ unsigned short lA[128 * 32];     // [m][k], row-major
  __shared__ unsigned short lB[128 * 32];     // [n][k], row-major (WT tile)
  const int tid  = threadIdx.x;
  const int wid  = tid >> 5, lane = tid & 31;
  const int half = lane >> 4, lm = lane & 15;
  const int wm0  = (wid & 1) * 64, wn0 = (wid >> 1) * 32;
  const int bm   = blockIdx.y * 128, bn = blockIdx.x * 128;

  v8f acc[4][2];
  #pragma unroll
  for (int mt = 0; mt < 4; ++mt)
    #pragma unroll
    for (int nt = 0; nt < 2; ++nt) acc[mt][nt] = vzero8();

  const int srow = tid >> 1, sseg = (tid & 1) * 16;  // 128 rows x 32 cols stage
  const unsigned short* aptr = A  + (size_t)(bm + srow) * K + sseg;
  const unsigned short* bptr = WT + (size_t)(bn + srow) * K + sseg;

  // register prefetch of tile kb=0
  v8us ra0 = *(const v8us*)aptr,      ra1 = *(const v8us*)(aptr + 8);
  v8us rb0 = *(const v8us*)bptr,      rb1 = *(const v8us*)(bptr + 8);

  for (int kb = 0; kb < K; kb += 32) {
    __syncthreads();
    *(v8us*)(&lA[srow * 32 + sseg])     = ra0;
    *(v8us*)(&lA[srow * 32 + sseg + 8]) = ra1;
    *(v8us*)(&lB[srow * 32 + sseg])     = rb0;
    *(v8us*)(&lB[srow * 32 + sseg + 8]) = rb1;
    __syncthreads();
    if (kb + 32 < K) {                        // prefetch next tile during math
      ra0 = *(const v8us*)(aptr + kb + 32);
      ra1 = *(const v8us*)(aptr + kb + 40);
      rb0 = *(const v8us*)(bptr + kb + 32);
      rb1 = *(const v8us*)(bptr + kb + 40);
    }
    v16bf afr[4], bfr[2];
    #pragma unroll
    for (int mt = 0; mt < 4; ++mt) { // A frag: K = 16h..16h+15 contiguous
      const unsigned short* p = &lA[(wm0 + mt * 16 + lm) * 32 + 16 * half];
      afr[mt] = usbf(combine8(*(const v8us*)p, *(const v8us*)(p + 8)));
    }
    #pragma unroll
    for (int nt = 0; nt < 2; ++nt) { // B frag: K = 16h..16h+15 contiguous
      bfr[nt] = usbf(*(const v16us*)(&lB[(wn0 + nt * 16 + lm) * 32 + 16 * half]));
    }
    #pragma unroll
    for (int mt = 0; mt < 4; ++mt)
      #pragma unroll
      for (int nt = 0; nt < 2; ++nt)
        acc[mt][nt] = wmma_bf16(afr[mt], bfr[nt], acc[mt][nt]);
  }

  // ---- epilogue: C-frag element (row = r+8h, col = lm) per VGPR r
  #pragma unroll
  for (int mt = 0; mt < 4; ++mt) {
    #pragma unroll
    for (int nt = 0; nt < 2; ++nt) {
      const int gn = bn + wn0 + nt * 16 + lm;
      const float bv = bias[gn];
      #pragma unroll
      for (int f = 0; f < 8; ++f) {
        const int gm = bm + wm0 + mt * 16 + f + 8 * half;
        float val = acc[mt][nt][f] + bv;
        if constexpr (EPI == 0) {             // QKV: M = B*T, N = 3*D
          const int bb = gm >> 11, t = gm & (Tc - 1);
          const int c = gn >> 10, rem = gn & (Dc - 1);
          const int h = rem >> 6, dh = rem & 63;
          if (c == 0)
            ou0[(((size_t)(bb * Hc + h) * Tc + t) << 6) + dh] = f2bf(val * QSCALE);
          else if (c == 1)
            ou1[(((size_t)(bb * Hc + h) * TSc + t) << 6) + dh] = f2bf(val);
          else                                // v: dh-major [b,h][dh][j]
            ou2[((size_t)(bb * Hc + h) * 64 + dh) * TSc + t] = f2bf(val);
        } else if constexpr (EPI == 1) {      // CKV: M = B*S, N = 2*D
          const int bb = gm >> 9, sI = gm & (Sc - 1);
          const int c = gn >> 10, rem = gn & (Dc - 1);
          const int h = rem >> 6, dh = rem & 63;
          const int j = Tc + sI;
          if (c == 0)
            ou1[(((size_t)(bb * Hc + h) * TSc + j) << 6) + dh] = f2bf(val);
          else                                // v: dh-major
            ou2[((size_t)(bb * Hc + h) * 64 + dh) * TSc + j] = f2bf(val);
        } else if constexpr (EPI == 2) {      // bias + exact GELU -> bf16
          val = 0.5f * val * (1.0f + erff(val * 0.70710678118654752440f));
          ou0[(size_t)gm * Fc + gn] = f2bf(val);
        } else {                              // bias + residual -> f32
          const size_t off = (size_t)gm * Dc + gn;
          of32[off] = val + addsrc[off];
        }
      }
    }
  }
}

// ---------------------------------------------------------------- attention
// Flash-style: block = 128 thr (4 waves), each wave owns 16 query rows and is
// fully independent (no barriers): K and V^T fragments come straight from
// global; only the per-wave P (C-layout -> A-layout) round-trips via LDS.
// Writes xres = x + attn_out.
__global__ __launch_bounds__(128)
void k_attn(const unsigned short* __restrict__ q, const unsigned short* __restrict__ kall,
            const unsigned short* __restrict__ vT, const float* __restrict__ x,
            float* __restrict__ xres) {
  __shared__ unsigned short ldsP[4][16 * 32]; // per-wave [row][key]
  const int tid  = threadIdx.x;
  const int wid  = tid >> 5, lane = tid & 31;
  const int half = lane >> 4, lm = lane & 15;
  const int q0 = blockIdx.x * 64;
  const int h  = blockIdx.y, b = blockIdx.z;
  const int i0 = q0 + wid * 16;

  const unsigned short* qp = q    + ((size_t)(b * Hc + h) * Tc)  * 64;
  const unsigned short* kp = kall + ((size_t)(b * Hc + h) * TSc) * 64;
  const unsigned short* vp = vT   + ((size_t)(b * Hc + h) * 64) * TSc; // [dh][j]

  // Q fragments (pre-scaled by 1/sqrt(dh)*log2e)
  v16bf qf[2];
  {
    const unsigned short* r = qp + (size_t)(i0 + lm) * 64;
    #pragma unroll
    for (int c = 0; c < 2; ++c) {
      v8us lo = *(const v8us*)(r + 32 * c + 16 * half);
      v8us hi = *(const v8us*)(r + 32 * c + 8 + 16 * half);
      qf[c] = usbf(combine8(lo, hi));
    }
  }

  v8f o[4];
  #pragma unroll
  for (int t = 0; t < 4; ++t) o[t] = vzero8();
  float mrow[8], lrow[8];
  #pragma unroll
  for (int f = 0; f < 8; ++f) { mrow[f] = -3.0e38f; lrow[f] = 0.0f; }

  // per-wave bounds: this wave's rows reach i0+15 -> last self block i0/32
  const int nSelf = (i0 >> 5) + 1;
  const int nBlk  = nSelf + (Sc >> 5);

  for (int bi = 0; bi < nBlk; ++bi) {
    const int j0 = (bi < nSelf) ? (bi << 5) : (Tc + ((bi - nSelf) << 5));
    // scores S = Q @ K^T : B-frag straight from global (contiguous dh per lane)
    v8f s[2];
    #pragma unroll
    for (int tn = 0; tn < 2; ++tn) {
      v8f a = vzero8();
      const unsigned short* kr = kp + (size_t)(j0 + tn * 16 + lm) * 64;
      #pragma unroll
      for (int c = 0; c < 2; ++c)
        a = wmma_bf16(qf[c], usbf(*(const v16us*)(kr + 32 * c + 16 * half)), a);
      s[tn] = a;
    }
    if (j0 < Tc) {                            // causal mask in self region
      #pragma unroll
      for (int tn = 0; tn < 2; ++tn)
        #pragma unroll
        for (int f = 0; f < 8; ++f) {
          const int j = j0 + tn * 16 + lm;
          const int i = i0 + f + 8 * half;
          if (j > i) s[tn][f] = -3.0e38f;
        }
    }
    // online softmax (exp2 domain); stats replicated across 16-lane half
    #pragma unroll
    for (int f = 0; f < 8; ++f) {
      float cm = fmaxf(s[0][f], s[1][f]);
      #pragma unroll
      for (int oo = 1; oo < 16; oo <<= 1) cm = fmaxf(cm, __shfl_xor(cm, oo, 32));
      const float mn = fmaxf(mrow[f], cm);
      const float alpha = exp2f(mrow[f] - mn);
      mrow[f] = mn;
      const float p0 = exp2f(s[0][f] - mn);
      const float p1 = exp2f(s[1][f] - mn);
      float ps = p0 + p1;
      #pragma unroll
      for (int oo = 1; oo < 16; oo <<= 1) ps += __shfl_xor(ps, oo, 32);
      lrow[f] = lrow[f] * alpha + ps;
      #pragma unroll
      for (int t = 0; t < 4; ++t) o[t][f] *= alpha;
      ldsP[wid][(f + 8 * half) * 32 + lm]      = f2bf(p0);
      ldsP[wid][(f + 8 * half) * 32 + 16 + lm] = f2bf(p1);
    }
    lds_fence_wave();                         // per-wave DS order + no reorder
    // O += P(16x32) @ V(32x64), V^T fragments direct from global
    v16bf pf;
    {
      const unsigned short* pr = &ldsP[wid][lm * 32 + 16 * half];
      pf = usbf(combine8(*(const v8us*)pr, *(const v8us*)(pr + 8)));
    }
    #pragma unroll
    for (int t = 0; t < 4; ++t) {
      v16bf vb = usbf(*(const v16us*)(vp + (size_t)(t * 16 + lm) * TSc + j0 + 16 * half));
      o[t] = wmma_bf16(pf, vb, o[t]);
    }
  }

  // finalize: xres = x + O / l
  float invl[8];
  #pragma unroll
  for (int f = 0; f < 8; ++f) invl[f] = 1.0f / lrow[f];
  #pragma unroll
  for (int t = 0; t < 4; ++t)
    #pragma unroll
    for (int f = 0; f < 8; ++f) {
      const int i = i0 + f + 8 * half;
      const size_t off = ((size_t)(b * Tc + i)) * Dc + h * 64 + t * 16 + lm;
      xres[off] = x[off] + o[t][f] * invl[f];
    }
}

// ---------------------------------------------------------------- launch
extern "C" void kernel_launch(void* const* d_in, const int* in_sizes, int n_in,
                              void* d_out, int out_size, void* d_ws, size_t ws_size,
                              hipStream_t stream) {
  const float* x     = (const float*)d_in[0];
  const float* ctx   = (const float*)d_in[1];
  const float* w_qkv = (const float*)d_in[2];
  const float* b_qkv = (const float*)d_in[3];
  const float* w_ckv = (const float*)d_in[4];
  const float* b_ckv = (const float*)d_in[5];
  const float* g1    = (const float*)d_in[6];
  const float* be1   = (const float*)d_in[7];
  const float* g2    = (const float*)d_in[8];
  const float* be2   = (const float*)d_in[9];
  const float* gf    = (const float*)d_in[10];
  const float* bfv   = (const float*)d_in[11];
  const float* w1    = (const float*)d_in[12];
  const float* b1    = (const float*)d_in[13];
  const float* w2    = (const float*)d_in[14];
  const float* b2    = (const float*)d_in[15];

  char* ws = (char*)d_ws;
  size_t off = 0;
  auto grab = [&](size_t bytes) -> char* {
    char* p = ws + off;
    off += (bytes + 255) & ~(size_t)255;
    return p;
  };
  unsigned short* wqkvT = (unsigned short*)grab((size_t)Dc * 3 * Dc * 2);
  unsigned short* wckvT = (unsigned short*)grab((size_t)Dc * 2 * Dc * 2);
  unsigned short* w1T   = (unsigned short*)grab((size_t)Dc * Fc * 2);
  unsigned short* w2T   = (unsigned short*)grab((size_t)Fc * Dc * 2);
  unsigned short* ctx_bf = (unsigned short*)grab((size_t)Bc * Sc * Dc * 2);
  unsigned short* ln1_bf = (unsigned short*)grab((size_t)Bc * Tc * Dc * 2);
  unsigned short* q_bf   = (unsigned short*)grab((size_t)Bc * Hc * Tc * 64 * 2);
  unsigned short* k_all  = (unsigned short*)grab((size_t)Bc * Hc * TSc * 64 * 2);
  unsigned short* vT_all = (unsigned short*)grab((size_t)Bc * Hc * TSc * 64 * 2);
  float*          xres   = (float*)grab((size_t)Bc * Tc * Dc * 4);
  unsigned short* lnff_bf = (unsigned short*)grab((size_t)Bc * Tc * Dc * 2);
  unsigned short* hdn_bf  = (unsigned short*)grab((size_t)Bc * Tc * Fc * 2);

  // weight convert + transpose: W[K][N] f32 -> WT[N][K] bf16
  k_wt_bf16<<<dim3(3 * Dc / 32, Dc / 32), 256, 0, stream>>>(w_qkv, wqkvT, Dc, 3 * Dc);
  k_wt_bf16<<<dim3(2 * Dc / 32, Dc / 32), 256, 0, stream>>>(w_ckv, wckvT, Dc, 2 * Dc);
  k_wt_bf16<<<dim3(Fc / 32, Dc / 32), 256, 0, stream>>>(w1, w1T, Dc, Fc);
  k_wt_bf16<<<dim3(Dc / 32, Fc / 32), 256, 0, stream>>>(w2, w2T, Fc, Dc);
  // context f32 -> bf16
  k_f32_to_bf16<<<1024, 256, 0, stream>>>(ctx, ctx_bf, Bc * Sc * Dc);

  // LN1(x) -> bf16
  k_ln1<<<Bc * Tc, 256, 0, stream>>>(x, g1, be1, ln1_bf);

  // QKV = LN1(x) @ w_qkv + b_qkv  -> scatter q(scaled)/k_self/v_self(dh-major)
  k_gemm<0><<<dim3(3 * Dc / 128, (Bc * Tc) / 128), 256, 0, stream>>>(
      ln1_bf, wqkvT, b_qkv, Bc * Tc, 3 * Dc, Dc, q_bf, k_all, vT_all, nullptr, nullptr);

  // CKV = context @ w_ckv + b_ckv -> scatter k_ctx/v_ctx (j = T + s)
  k_gemm<1><<<dim3(2 * Dc / 128, (Bc * Sc) / 128), 256, 0, stream>>>(
      ctx_bf, wckvT, b_ckv, Bc * Sc, 2 * Dc, Dc, nullptr, k_all, vT_all, nullptr, nullptr);

  // attention + residual: xres = x + attn_out
  k_attn<<<dim3(Tc / 64, Hc, Bc), 128, 0, stream>>>(q_bf, k_all, vT_all, x, xres);

  // LN(LN(xres; g2, be2); gf, bf) -> bf16
  k_ln2<<<Bc * Tc, 256, 0, stream>>>(xres, g2, be2, gf, bfv, lnff_bf);

  // FFN1: GELU(lnff @ w1 + b1) -> bf16
  k_gemm<2><<<dim3(Fc / 128, (Bc * Tc) / 128), 256, 0, stream>>>(
      lnff_bf, w1T, b1, Bc * Tc, Fc, Dc, hdn_bf, nullptr, nullptr, nullptr, nullptr);

  // FFN2: out = xres + (hdn @ w2 + b2)
  k_gemm<3><<<dim3(Dc / 128, (Bc * Tc) / 128), 256, 0, stream>>>(
      hdn_bf, w2T, b2, Bc * Tc, Dc, Fc, nullptr, nullptr, nullptr, xres, (float*)d_out);
}
